// AttentionGroup_5231270166613
// MI455X (gfx1250) — matrix operand order repaired and checked by reading, not compiled, not run on hardware
//
#include <hip/hip_runtime.h>
#include <cstdint>

// Problem constants (from the reference)
#define BB 4
#define SS 2
#define PP 1024   // seq
#define DD 1024   // d_embed
#define II 16     // heads
#define HH 64     // head dim
#define FF (II*HH)  // 1024
#define NBS (BB*SS) // 8

typedef __attribute__((ext_vector_type(16))) __bf16         v16bf;
typedef __attribute__((ext_vector_type(8)))  float          v8f;
typedef __attribute__((ext_vector_type(16))) unsigned short v16u;
typedef __attribute__((ext_vector_type(8)))  unsigned short v8u;
typedef __attribute__((ext_vector_type(4)))  float          vf4;
typedef __attribute__((ext_vector_type(4)))  int            v4i;

// Pointer-to-AS1/AS3 int4 (pointee carries the address space)
typedef __attribute__((address_space(1))) v4i* as1_v4i_ptr;
typedef __attribute__((address_space(3))) v4i* as3_v4i_ptr;

// CDNA5 async global->LDS copy (ASYNCcnt-tracked), with safe fallback.
#if defined(__has_builtin)
#if __has_builtin(__builtin_amdgcn_global_load_async_to_lds_b128)
#define HAVE_ASYNC_LDS 1
#endif
#endif
#ifndef HAVE_ASYNC_LDS
#define HAVE_ASYNC_LDS 0
#endif

__device__ __forceinline__ void copy16B_to_lds(const unsigned short* src,
                                               unsigned short* lds_dst) {
#if HAVE_ASYNC_LDS
  __builtin_amdgcn_global_load_async_to_lds_b128(
      (as1_v4i_ptr)(uintptr_t)src,
      (as3_v4i_ptr)(uintptr_t)lds_dst, 0, 0);
#else
  *(v8u*)lds_dst = *(const v8u*)src;
#endif
}

__device__ __forceinline__ void wait_async_copies() {
#if HAVE_ASYNC_LDS
#if defined(__has_builtin) && __has_builtin(__builtin_amdgcn_s_wait_asynccnt)
  __builtin_amdgcn_s_wait_asynccnt(0);
#else
  asm volatile("s_wait_asynccnt 0x0" ::: "memory");
#endif
#endif
}

// fp32 -> bf16 (round to nearest even) — used in the memory-bound pre-pass,
// for the small softmax-prob tiles and for the bf16 Q/K/V and Z output stores.
__device__ __forceinline__ unsigned short f2bf(float f) {
  union { float f; unsigned u; } v; v.f = f;
  unsigned r = v.u + 0x7FFFu + ((v.u >> 16) & 1u);
  return (unsigned short)(r >> 16);
}

// D = A(16x32 bf16) * B(32x16 bf16) + C(f32), wave32 WMMA
__device__ __forceinline__ v8f wmma_bf16(v16u a, v16u b, v8f c) {
  return __builtin_amdgcn_wmma_f32_16x16x32_bf16(
      false, __builtin_bit_cast(v16bf, a),
      false, __builtin_bit_cast(v16bf, b),
      (short)0, c, false, false);
}

// ---------------------------------------------------------------------------
// Kernel 0: bulk fp32 -> bf16 conversion (memory bound; ~4us at 23.3 TB/s)
// ---------------------------------------------------------------------------
__global__ __launch_bounds__(256)
void cvt_bf16_kernel(const float* __restrict__ src,
                     unsigned short* __restrict__ dst, int n) {
  int i = (int)(blockIdx.x * blockDim.x + threadIdx.x) * 8;
  if (i >= n) return;
  vf4 a = *(const vf4*)(src + i);
  vf4 b = *(const vf4*)(src + i + 4);
  v8u o;
  o[0] = f2bf(a[0]); o[1] = f2bf(a[1]); o[2] = f2bf(a[2]); o[3] = f2bf(a[3]);
  o[4] = f2bf(b[0]); o[5] = f2bf(b[1]); o[6] = f2bf(b[2]); o[7] = f2bf(b[3]);
  *(v8u*)(dst + i) = o;
}

// ---------------------------------------------------------------------------
// Kernel 1: Q/K/V projections.  C[p,f] = sum_d X[p,d] * W[f,d]
// Block = 8 waves sharing one (mat,bs,nt): wave w owns a 32(M=p) x 64(N=f) tile
// (two A row-blocks x 4 accumульators each -> 8 WMMAs per B tile).
// B tile (64 f-rows x 32 k) double-buffered in LDS via async copies.
// Q,K stored bf16 [bs,i,p,h]; V stored transposed [bs,i,h,p].
// ---------------------------------------------------------------------------
__global__ __launch_bounds__(256)
void qkv_proj_kernel(const unsigned short* __restrict__ Xb,
                     const unsigned short* __restrict__ Wqb,
                     const unsigned short* __restrict__ Wkb,
                     const unsigned short* __restrict__ Wvb,
                     unsigned short* __restrict__ Qb,
                     unsigned short* __restrict__ Kb,
                     unsigned short* __restrict__ Vt) {
  __shared__ __align__(64) unsigned short Bsh[2][64 * 32];  // 2 x 4KB

  int blk = blockIdx.x;
  int nt  = blk & 15; blk >>= 4;   // 16 n-tiles of 64 f-cols
  int ptg = blk & 3;  blk >>= 2;   // 4 groups of 8 p-tiles (32 rows each)
  int bs  = blk & 7;  blk >>= 3;   // 8 (b,s)
  int mat = blk;                    // 0=Q 1=K 2=V
  int s = bs & 1;

  int tid  = threadIdx.x;
  int lane = tid & 31;
  int wave = tid >> 5;
  int hl = lane >> 4, l16 = lane & 15;
  int n0 = nt * 64;
  int p0 = (ptg * 8 + wave) * 32;

  const unsigned short* W =
      ((mat == 0) ? Wqb : (mat == 1) ? Wkb : Wvb) + (size_t)s * FF * DD;
  const unsigned short* xr0 = Xb + (size_t)((size_t)bs * PP + p0 + l16) * DD;
  const unsigned short* xr1 = xr0 + (size_t)16 * DD;

  // staging: thread t copies 16B: row = t>>2 (0..63), quarter = t&3
  int srow = tid >> 2, sq4 = (tid & 3) * 8;
  const unsigned short* wsrc = W + (size_t)(n0 + srow) * DD + sq4;

  copy16B_to_lds(wsrc + 0, &Bsh[0][srow * 32 + sq4]);

  v8f acc[2][4] = {};
  for (int k0 = 0; k0 < DD; k0 += 32) {
    int buf = (k0 >> 5) & 1;
    wait_async_copies();
    __syncthreads();
    if (k0 + 32 < DD)
      copy16B_to_lds(wsrc + k0 + 32, &Bsh[buf ^ 1][srow * 32 + sq4]);

    __builtin_prefetch(xr0 + k0 + 256, 0, 0);
    // A operands (bf16): k = k0+16*hl+j
    v8u a0lo = *(const v8u*)(xr0 + k0 + 16 * hl);
    v8u a0hi = *(const v8u*)(xr0 + k0 + 16 * hl + 8);
    v8u a1lo = *(const v8u*)(xr1 + k0 + 16 * hl);
    v8u a1hi = *(const v8u*)(xr1 + k0 + 16 * hl + 8);
    v16u au0, au1;
#pragma unroll
    for (int j = 0; j < 8; ++j) {
      au0[j] = a0lo[j]; au0[j + 8] = a0hi[j];
      au1[j] = a1lo[j]; au1[j + 8] = a1hi[j];
    }
#pragma unroll
    for (int t = 0; t < 4; ++t) {
      // B operand from LDS: col f-local = 16t+l16; k elems = 16*hl + j
      v16u bu = *(const v16u*)(&Bsh[buf][(16 * t + l16) * 32 + 16 * hl]);
      acc[0][t] = wmma_bf16(au0, bu, acc[0][t]);
      acc[1][t] = wmma_bf16(au1, bu, acc[1][t]);
    }
    __syncthreads();  // all reads of Bsh[buf] done before it is refilled
  }

  // C layout: element (M = r + 8*hl, N = l16) in acc[mr][t][r]
#pragma unroll
  for (int mr = 0; mr < 2; ++mr)
#pragma unroll
    for (int t = 0; t < 4; ++t) {
      int f = n0 + 16 * t + l16;
      int i = f >> 6, h = f & 63;
#pragma unroll
      for (int r = 0; r < 8; ++r) {
        int p = p0 + 16 * mr + r + 8 * hl;
        unsigned short val = f2bf(acc[mr][t][r]);
        if (mat == 0)      Qb[(((size_t)bs * II + i) * PP + p) * HH + h] = val;
        else if (mat == 1) Kb[(((size_t)bs * II + i) * PP + p) * HH + h] = val;
        else               Vt[(((size_t)bs * II + i) * HH + h) * PP + p] = val;
      }
    }
}

// ---------------------------------------------------------------------------
// Kernel 2: causal flash attention per (bs, head, 16-query tile).
//   logits = (p<=q) ? score/8 : -1.25e9   (== (s*mask - 1e10*(1-mask))/sqrt(64))
//   online softmax in f32; O += P(bf16 via LDS relayout) x V(bf16)
// Writes z_flat bf16 as [bs, q, i*H + h].
// ---------------------------------------------------------------------------
__global__ __launch_bounds__(256)
void attn_kernel(const unsigned short* __restrict__ Qb,
                 const unsigned short* __restrict__ Kb,
                 const unsigned short* __restrict__ Vt,
                 unsigned short* __restrict__ Z) {
  __shared__ unsigned short ldsP[8 * 512];  // per-wave 16x32 bf16 prob tile
  int wfid = (int)((blockIdx.x * blockDim.x + threadIdx.x) >> 5);
  int lane = threadIdx.x & 31;
  int hl = lane >> 4, l16 = lane & 15;
  unsigned short* myP = ldsP + ((threadIdx.x >> 5) * 512);

  int qt = wfid & 63; wfid >>= 6;
  int ih = wfid & 15; wfid >>= 4;
  int bs = wfid;
  int q0 = qt * 16;
  size_t head = (size_t)bs * II + ih;
  const unsigned short* Q = Qb + head * PP * HH;
  const unsigned short* K = Kb + head * PP * HH;
  const unsigned short* V = Vt + head * HH * PP;

  // Q as A operand, two K-chunks: h in [0,32) and [32,64)
  const unsigned short* qrow = Q + (size_t)(q0 + l16) * HH;
  v16u aq0, aq1;
  {
    v8u t0 = *(const v8u*)(qrow + 16 * hl);
    v8u t1 = *(const v8u*)(qrow + 16 * hl + 8);
    v8u t2 = *(const v8u*)(qrow + 32 + 16 * hl);
    v8u t3 = *(const v8u*)(qrow + 32 + 16 * hl + 8);
#pragma unroll
    for (int j = 0; j < 8; ++j) {
      aq0[j] = t0[j]; aq0[j + 8] = t1[j];
      aq1[j] = t2[j]; aq1[j + 8] = t3[j];
    }
  }

  v8f o[4] = {};
  float m[8], l[8];
#pragma unroll
  for (int r = 0; r < 8; ++r) { m[r] = -3.0e38f; l[r] = 0.0f; }

  for (int p0 = 0; p0 <= q0 + 15; p0 += 32) {
    float lg[2][8];
#pragma unroll
    for (int u = 0; u < 2; ++u) {
      int pc = p0 + 16 * u + l16;  // this lane's key column
      const unsigned short* krow = K + (size_t)pc * HH;
      v16u b0 = *(const v16u*)(krow + 16 * hl);        // h = 16*hl + j
      v16u b1 = *(const v16u*)(krow + 32 + 16 * hl);   // h = 32 + 16*hl + j
      v8f sc = {};
      sc = wmma_bf16(aq0, b0, sc);
      sc = wmma_bf16(aq1, b1, sc);
#pragma unroll
      for (int r = 0; r < 8; ++r) {
        int qg = q0 + r + 8 * hl;
        lg[u][r] = (pc <= qg) ? sc[r] * 0.125f : -1.25e9f;
      }
    }

    // per-row online softmax stats (row M spread over 16 lanes of this half)
    float mn[8], corr[8], pr[2][8];
#pragma unroll
    for (int r = 0; r < 8; ++r) {
      float v = fmaxf(lg[0][r], lg[1][r]);
      v = fmaxf(v, __shfl_xor(v, 1, 32));
      v = fmaxf(v, __shfl_xor(v, 2, 32));
      v = fmaxf(v, __shfl_xor(v, 4, 32));
      v = fmaxf(v, __shfl_xor(v, 8, 32));
      mn[r] = fmaxf(m[r], v);
      corr[r] = __expf(m[r] - mn[r]);
      pr[0][r] = __expf(lg[0][r] - mn[r]);
      pr[1][r] = __expf(lg[1][r] - mn[r]);
      float srow = pr[0][r] + pr[1][r];
      srow += __shfl_xor(srow, 1, 32);
      srow += __shfl_xor(srow, 2, 32);
      srow += __shfl_xor(srow, 4, 32);
      srow += __shfl_xor(srow, 8, 32);
      l[r] = l[r] * corr[r] + srow;
      m[r] = mn[r];
    }
#pragma unroll
    for (int t = 0; t < 4; ++t)
#pragma unroll
      for (int r = 0; r < 8; ++r) o[t][r] *= corr[r];

    // re-layout P (C layout -> A layout) through LDS, row-major 16x32 bf16
#pragma unroll
    for (int u = 0; u < 2; ++u)
#pragma unroll
      for (int r = 0; r < 8; ++r)
        myP[(r + 8 * hl) * 32 + 16 * u + l16] = f2bf(pr[u][r]);
    asm volatile("" ::: "memory");  // keep DS store->load program order (HW in-order)

    v16u ap;
    {
      const unsigned short* prow = myP + l16 * 32;
      v8u plo = *(const v8u*)(prow + 16 * hl);
      v8u phi = *(const v8u*)(prow + 16 * hl + 8);
#pragma unroll
      for (int j = 0; j < 8; ++j) { ap[j] = plo[j]; ap[j + 8] = phi[j]; }
    }

    // O[qtile, htile t] += P(16x32) x V^T-slice(32x16)
#pragma unroll
    for (int t = 0; t < 4; ++t) {
      const unsigned short* vrow = V + (size_t)(16 * t + l16) * PP + p0 + 16 * hl;
      v16u bv = *(const v16u*)vrow;
      o[t] = wmma_bf16(ap, bv, o[t]);
    }
  }

  // normalize and write z_flat [bs, q, i*H + h] as bf16
#pragma unroll
  for (int r = 0; r < 8; ++r) l[r] = 1.0f / l[r];
#pragma unroll
  for (int t = 0; t < 4; ++t)
#pragma unroll
    for (int r = 0; r < 8; ++r) {
      int q = q0 + r + 8 * hl;
      int f = ih * HH + 16 * t + l16;
      Z[((size_t)bs * PP + q) * FF + f] = f2bf(o[t][r] * l[r]);
    }
}

// ---------------------------------------------------------------------------
// Kernel 3: out[q,d] = sum_f z[q,f] * W_O[d,f]   (per bs; fp32 output)
// Same cooperative B-staging; wave owns 32(q) x 64(d).
// ---------------------------------------------------------------------------
__global__ __launch_bounds__(256)
void out_proj_kernel(const unsigned short* __restrict__ Z,
                     const unsigned short* __restrict__ Wob,
                     float* __restrict__ out) {
  __shared__ __align__(64) unsigned short Bsh[2][64 * 32];

  int blk = blockIdx.x;
  int nt  = blk & 15; blk >>= 4;   // 16 n-tiles of 64 d-cols
  int qtg = blk & 3;  blk >>= 2;   // 4 groups of 8 q-tiles (32 rows each)
  int bs  = blk;                    // 8 (b,s)
  int s = bs & 1;

  int tid  = threadIdx.x;
  int lane = tid & 31;
  int wave = tid >> 5;
  int hl = lane >> 4, l16 = lane & 15;
  int n0 = nt * 64;
  int q0 = (qtg * 8 + wave) * 32;

  const unsigned short* W = Wob + (size_t)s * DD * FF;
  const unsigned short* zr0 = Z + ((size_t)bs * PP + q0 + l16) * FF;
  const unsigned short* zr1 = zr0 + (size_t)16 * FF;

  int srow = tid >> 2, sq4 = (tid & 3) * 8;
  const unsigned short* wsrc = W + (size_t)(n0 + srow) * FF + sq4;

  copy16B_to_lds(wsrc + 0, &Bsh[0][srow * 32 + sq4]);

  v8f acc[2][4] = {};
  for (int k0 = 0; k0 < FF; k0 += 32) {
    int buf = (k0 >> 5) & 1;
    wait_async_copies();
    __syncthreads();
    if (k0 + 32 < FF)
      copy16B_to_lds(wsrc + k0 + 32, &Bsh[buf ^ 1][srow * 32 + sq4]);

    __builtin_prefetch(zr0 + k0 + 256, 0, 0);
    v8u z0lo = *(const v8u*)(zr0 + k0 + 16 * hl);
    v8u z0hi = *(const v8u*)(zr0 + k0 + 16 * hl + 8);
    v8u z1lo = *(const v8u*)(zr1 + k0 + 16 * hl);
    v8u z1hi = *(const v8u*)(zr1 + k0 + 16 * hl + 8);
    v16u au0, au1;
#pragma unroll
    for (int j = 0; j < 8; ++j) {
      au0[j] = z0lo[j]; au0[j + 8] = z0hi[j];
      au1[j] = z1lo[j]; au1[j + 8] = z1hi[j];
    }
#pragma unroll
    for (int t = 0; t < 4; ++t) {
      v16u bu = *(const v16u*)(&Bsh[buf][(16 * t + l16) * 32 + 16 * hl]);
      acc[0][t] = wmma_bf16(au0, bu, acc[0][t]);
      acc[1][t] = wmma_bf16(au1, bu, acc[1][t]);
    }
    __syncthreads();
  }
#pragma unroll
  for (int mr = 0; mr < 2; ++mr)
#pragma unroll
    for (int t = 0; t < 4; ++t)
#pragma unroll
      for (int r = 0; r < 8; ++r) {
        int q = q0 + 16 * mr + r + 8 * hl;
        int d = n0 + 16 * t + l16;
        out[((size_t)bs * PP + q) * DD + d] = acc[mr][t][r];
      }
}

// ---------------------------------------------------------------------------
extern "C" void kernel_launch(void* const* d_in, const int* in_sizes, int n_in,
                              void* d_out, int out_size, void* d_ws, size_t ws_size,
                              hipStream_t stream) {
  const float* x  = (const float*)d_in[0];
  const float* Wk = (const float*)d_in[1];
  const float* Wq = (const float*)d_in[2];
  const float* Wv = (const float*)d_in[3];
  const float* Wo = (const float*)d_in[4];

  // Workspace (bf16 as ushort):
  //   Qb,Kb [bs,i,p,h]  Vt [bs,i,h,p]  Xbf [bs,p,d] (aliased by Z [bs,q,i*H])
  //   Wqb/Wkb/Wvb [s,f,d]  Wob [s,d,f]        total 80 MB
  const size_t E_BIG = (size_t)NBS * II * PP * HH;  // 8M
  const size_t E_W   = (size_t)SS * FF * DD;        // 2M
  unsigned short* Qb  = (unsigned short*)d_ws;
  unsigned short* Kb  = Qb + E_BIG;
  unsigned short* Vt  = Kb + E_BIG;
  unsigned short* Xbf = Vt + E_BIG;
  unsigned short* Zb  = Xbf;          // alias: X dead after projections
  unsigned short* Wqb = Xbf + E_BIG;
  unsigned short* Wkb = Wqb + E_W;
  unsigned short* Wvb = Wkb + E_W;
  unsigned short* Wob = Wvb + E_W;

  float* out = (float*)d_out;

  // fp32 -> bf16 pre-pass (memory bound)
  cvt_bf16_kernel<<<(int)(E_BIG / 8 / 256), 256, 0, stream>>>(x,  Xbf, (int)E_BIG);
  cvt_bf16_kernel<<<(int)(E_W  / 8 / 256), 256, 0, stream>>>(Wq, Wqb, (int)E_W);
  cvt_bf16_kernel<<<(int)(E_W  / 8 / 256), 256, 0, stream>>>(Wk, Wkb, (int)E_W);
  cvt_bf16_kernel<<<(int)(E_W  / 8 / 256), 256, 0, stream>>>(Wv, Wvb, (int)E_W);
  cvt_bf16_kernel<<<(int)(E_W  / 8 / 256), 256, 0, stream>>>(Wo, Wob, (int)E_W);

  // 3 mats * 8 bs * 4 p-groups * 16 n-tiles = 1536 blocks (8 waves each)
  qkv_proj_kernel<<<1536, 256, 0, stream>>>(Xbf, Wqb, Wkb, Wvb, Qb, Kb, Vt);
  // 8*16*64 = 8192 waves
  attn_kernel<<<1024, 256, 0, stream>>>(Qb, Kb, Vt, Zb);
  // 8 bs * 4 q-groups * 16 n-tiles = 512 blocks
  out_proj_kernel<<<512, 256, 0, stream>>>(Zb, Wob, out);
}
